// CrissCrossAttention_670014898439
// MI455X (gfx1250) — matrix operand order repaired and checked by reading, not compiled, not run on hardware
//
#include <hip/hip_runtime.h>

#define B_  8
#define C_  512
#define CQ_ 64
#define H_  96
#define W_  96
#define N_  (H_*W_)   // 9216
#define J_  (2*H_)    // 192

typedef __bf16 bf16_t;
typedef __attribute__((ext_vector_type(16))) __bf16 v16bf;
typedef __attribute__((ext_vector_type(8)))  float  v8f;
typedef __attribute__((ext_vector_type(4)))  unsigned int u32x4;
typedef __attribute__((ext_vector_type(2)))  unsigned int u32x2;

union V16U { v16bf v; u32x4 q[2]; };

__device__ __forceinline__ unsigned pack_bf16(float lo, float hi) {
  union { __bf16 h[2]; unsigned u; } p;
  p.h[0] = (__bf16)lo;
  p.h[1] = (__bf16)hi;
  return p.u;
}

// A operand (16x32 bf16): lane holds row m = m0 + lane%16.
// element e -> k = e + (e<8 ? 0 : 8) + (lane/16)*8  => two 16B runs at +0 and +16.
__device__ __forceinline__ v16bf lds_load_a(const bf16_t* base, int stride, int m0, int kk, int lane) {
  const bf16_t* row = base + (size_t)(m0 + (lane & 15)) * stride + kk + ((lane >> 4) << 3);
  V16U u;
  u.q[0] = *(const u32x4*)(row);
  u.q[1] = *(const u32x4*)(row + 16);
  return u.v;
}

// B operand (32x16 bf16), LDS stored [n][k]: lane holds col n = n0 + lane%16,
// element e -> k = e + (e<8 ? 0 : 8) + (lane/16)*8  => two 16B runs at +0 and +16.
__device__ __forceinline__ v16bf lds_load_b(const bf16_t* base, int stride, int n0, int kk, int lane) {
  const bf16_t* row = base + (size_t)(n0 + (lane & 15)) * stride + kk + ((lane >> 4) << 3);
  V16U u;
  u.q[0] = *(const u32x4*)(row);
  u.q[1] = *(const u32x4*)(row + 16);
  return u.v;
}

__device__ __forceinline__ v8f wmma_bf16(v16bf a, v16bf b, v8f c) {
  return __builtin_amdgcn_wmma_f32_16x16x32_bf16(false, a, false, b, (short)0, c, false, false);
}

// ---------------- Kernel 1: pack weights + biases to bf16/f32 ----------------
__global__ void k_pack(const float* __restrict__ Wq, const float* __restrict__ bq,
                       const float* __restrict__ Wk, const float* __restrict__ bk,
                       const float* __restrict__ Wv, const float* __restrict__ bv,
                       bf16_t* __restrict__ Wall, float* __restrict__ ball) {
  int i = blockIdx.x * 256 + threadIdx.x;
  if (i < 640 * C_) {
    int r = i / C_, c = i - r * C_;
    float v;
    if (r < CQ_)          v = Wq[r * C_ + c];
    else if (r < 2*CQ_)   v = Wk[(r - CQ_) * C_ + c];
    else                  v = Wv[(r - 2*CQ_) * C_ + c];
    Wall[i] = (bf16_t)v;
  }
  if (i < 640) {
    float bb;
    if (i < CQ_)          bb = bq[i];
    else if (i < 2*CQ_)   bb = bk[i - CQ_];
    else                  bb = bv[i - 2*CQ_];
    ball[i] = bb;
  }
}

// ---------------- Kernel 2: fused QKV projection GEMM (WMMA bf16) ------------
// O(640 x 9216) = Wall(640x512) * X_b(512x9216) + bias, rows split to q/k/v.
__global__ __launch_bounds__(128)
void k_qkv(const float* __restrict__ x, const bf16_t* __restrict__ Wall,
           const float* __restrict__ ball,
           bf16_t* __restrict__ qb, bf16_t* __restrict__ kb, bf16_t* __restrict__ vb) {
  __shared__ __attribute__((aligned(16))) bf16_t As[64][32];   // [m][k]
  __shared__ __attribute__((aligned(16))) bf16_t Bs[64][32];   // [n][k]
  const int tid = threadIdx.x, lane = tid & 31, wv = tid >> 5;
  const int colBase = blockIdx.x * 64;
  const int rowBase = blockIdx.y * 64;
  const int b = blockIdx.z;
  const float* xb = x + (size_t)b * C_ * N_;

  v8f acc[4];
  #pragma unroll
  for (int t = 0; t < 4; ++t) for (int e = 0; e < 8; ++e) acc[t][e] = 0.f;

  for (int kk = 0; kk < C_; kk += 32) {
    // weights (bf16): 64x32 = 256 x b128; 2 per thread
    #pragma unroll
    for (int it = 0; it < 2; ++it) {
      int idx = tid + it * 128;                 // 0..255
      int r = idx >> 2, c8 = (idx & 3) << 3;
      *(u32x4*)&As[r][c8] = *(const u32x4*)(Wall + (size_t)(rowBase + r) * C_ + kk + c8);
    }
    // x (f32 -> bf16): pack k,k+1 per n; 1024 pairs; 8 per thread; b32 LDS stores
    #pragma unroll
    for (int it = 0; it < 8; ++it) {
      int p = tid + it * 128;                   // 0..1023
      int k = (p >> 6) << 1, n = p & 63;
      const float* src = xb + (size_t)(kk + k) * N_ + colBase + n;
      *(unsigned*)&Bs[n][k] = pack_bf16(src[0], src[N_]);
    }
    if (kk + 32 < C_)
      __builtin_prefetch(&xb[(size_t)(kk + 32 + (tid >> 6)) * N_ + colBase + (tid & 63)], 0, 1);
    __syncthreads();
    v16bf a = lds_load_a(&As[0][0], 32, wv * 16, 0, lane);
    #pragma unroll
    for (int t = 0; t < 4; ++t) {
      v16bf bm = lds_load_b(&Bs[0][0], 32, t * 16, 0, lane);
      acc[t] = wmma_bf16(a, bm, acc[t]);
    }
    __syncthreads();
  }

  const int half = lane >> 4, n_lo = lane & 15;
  #pragma unroll
  for (int t = 0; t < 4; ++t) {
    int n = colBase + t * 16 + n_lo;
    #pragma unroll
    for (int r = 0; r < 8; ++r) {
      int row = rowBase + wv * 16 + r + half * 8;
      bf16_t o = (bf16_t)(acc[t][r] + ball[row]);
      if (row < CQ_)        qb[((size_t)b * CQ_ + row) * N_ + n] = o;
      else if (row < 2*CQ_) kb[((size_t)b * CQ_ + (row - CQ_)) * N_ + n] = o;
      else                  vb[((size_t)b * C_  + (row - 2*CQ_)) * N_ + n] = o;
    }
  }
}

// ---------------- Kernel 3/4: energies (WMMA, 96x96, k=64) -------------------
// MODE 0: fixed w, E_H[h,g] = sum_c q[c,h,w] k[c,g,w]  -> e[b,h,w,g]
// MODE 1: fixed h, E_W[w,v] = sum_c q[c,h,w] k[c,h,v]  -> e[b,h,w,96+v]
template<int MODE>
__global__ __launch_bounds__(192)
void k_energy(const bf16_t* __restrict__ qb, const bf16_t* __restrict__ kb,
              float* __restrict__ e) {
  __shared__ __attribute__((aligned(16))) bf16_t Aq[96][64];   // [m][c]
  __shared__ __attribute__((aligned(16))) bf16_t Bk[96][64];   // [n][c]
  const int tid = threadIdx.x, lane = tid & 31, wv = tid >> 5;
  const int fix = blockIdx.x;       // w (MODE 0) or h (MODE 1)
  const int b = blockIdx.y;
  const bf16_t* qB = qb + (size_t)b * CQ_ * N_;
  const bf16_t* kB = kb + (size_t)b * CQ_ * N_;

  if (MODE == 1) {
    // m is contiguous in global: 8B (4xbf16) loads, scattered LDS stores
    #pragma unroll
    for (int it = 0; it < 8; ++it) {
      int qd = tid + it * 192;                  // 0..1535
      int c = qd / 24, m4 = (qd % 24) * 4;
      size_t off = (size_t)c * N_ + (size_t)fix * W_ + m4;
      union { u32x2 v; bf16_t h[4]; } aq, bk_;
      aq.v  = *(const u32x2*)(qB + off);
      bk_.v = *(const u32x2*)(kB + off);
      #pragma unroll
      for (int j = 0; j < 4; ++j) {
        Aq[m4 + j][c] = aq.h[j];
        Bk[m4 + j][c] = bk_.h[j];
      }
    }
  } else {
    for (int i = tid; i < 96*64; i += 192) {
      int m = i % 96, c = i / 96;
      size_t off = (size_t)c * N_ + (size_t)m * W_ + fix;
      Aq[m][c] = qB[off];
      Bk[m][c] = kB[off];
    }
  }
  __syncthreads();

  const int m0 = wv * 16;
  v8f acc[6];
  #pragma unroll
  for (int t = 0; t < 6; ++t) for (int r = 0; r < 8; ++r) acc[t][r] = 0.f;
  #pragma unroll
  for (int kk = 0; kk < 64; kk += 32) {
    v16bf a = lds_load_a(&Aq[0][0], 64, m0, kk, lane);
    #pragma unroll
    for (int t = 0; t < 6; ++t) {
      v16bf bb = lds_load_b(&Bk[0][0], 64, t * 16, kk, lane);
      acc[t] = wmma_bf16(a, bb, acc[t]);
    }
  }
  const int half = lane >> 4, n_lo = lane & 15;
  #pragma unroll
  for (int t = 0; t < 6; ++t) {
    int n = t * 16 + n_lo;
    #pragma unroll
    for (int r = 0; r < 8; ++r) {
      int m = m0 + r + half * 8;
      size_t idx = (MODE == 0)
        ? ((((size_t)b * H_ + m) * W_ + fix) * J_ + n)
        : ((((size_t)b * H_ + fix) * W_ + m) * J_ + H_ + n);
      e[idx] = acc[t][r];
    }
  }
}

// ---------------- Kernel 5: softmax over axis h (per (b,w,j) column) ---------
__global__ __launch_bounds__(192)
void k_softmax(const float* __restrict__ e, bf16_t* __restrict__ attn) {
  const int j = threadIdx.x;            // 0..191
  const int w = blockIdx.x, b = blockIdx.y;
  const size_t base = (((size_t)b * H_) * W_ + w) * J_ + j;
  const size_t sh = (size_t)W_ * J_;
  float mx = -3.4e38f;
  for (int h = 0; h < H_; ++h) mx = fmaxf(mx, e[base + h * sh]);
  float sum = 0.f;
  for (int h = 0; h < H_; ++h) sum += __expf(e[base + h * sh] - mx);
  const float inv = 1.f / sum;
  for (int h = 0; h < H_; ++h)
    attn[base + h * sh] = (bf16_t)(__expf(e[base + h * sh] - mx) * inv);
}

// ---------------- Kernel 6: out_H (per (b,w), WMMA 128x96, k=96) -------------
__global__ __launch_bounds__(256)
void k_outH(const bf16_t* __restrict__ vb, const bf16_t* __restrict__ attn,
            float* __restrict__ outH) {
  __shared__ __attribute__((aligned(16))) bf16_t Av[128][96];  // [c][g]
  __shared__ __attribute__((aligned(16))) bf16_t Ba[96][96];   // [h][g]
  const int tid = threadIdx.x, lane = tid & 31, wv = tid >> 5;
  const int w = blockIdx.x, cBase = blockIdx.y * 128, b = blockIdx.z;
  const bf16_t* vB = vb + (size_t)b * C_ * N_;

  // V tile: strided gather (stride W_) -- per element
  for (int i = tid; i < 128*96; i += 256) {
    int g = i % 96, ci = i / 96;
    Av[ci][g] = vB[(size_t)(cBase + ci) * N_ + (size_t)g * W_ + w];
  }
  // attn tile: g-contiguous -> 8B loads + 8B LDS stores
  #pragma unroll
  for (int it = 0; it < 9; ++it) {
    int qd = tid + it * 256;                    // 0..2303
    int h = qd / 24, g4 = (qd % 24) * 4;
    *(u32x2*)&Ba[h][g4] =
      *(const u32x2*)(attn + ((((size_t)b * H_ + h) * W_ + w) * J_ + g4));
  }
  __syncthreads();

  const int m0 = wv * 16;
  v8f acc[6];
  #pragma unroll
  for (int t = 0; t < 6; ++t) for (int r = 0; r < 8; ++r) acc[t][r] = 0.f;
  #pragma unroll
  for (int kk = 0; kk < 96; kk += 32) {
    v16bf a = lds_load_a(&Av[0][0], 96, m0, kk, lane);
    #pragma unroll
    for (int t = 0; t < 6; ++t) {
      v16bf bb = lds_load_b(&Ba[0][0], 96, t * 16, kk, lane);
      acc[t] = wmma_bf16(a, bb, acc[t]);
    }
  }
  const int half = lane >> 4, n_lo = lane & 15;
  #pragma unroll
  for (int t = 0; t < 6; ++t) {
    int h = t * 16 + n_lo;
    #pragma unroll
    for (int r = 0; r < 8; ++r) {
      int c = cBase + m0 + r + half * 8;
      outH[(((size_t)b * C_ + c) * H_ + h) * W_ + w] = acc[t][r];
    }
  }
}

// ---------------- Kernel 7: out_W + epilogue gamma*(outH+outW)+x -------------
__global__ __launch_bounds__(256)
void k_outW(const bf16_t* __restrict__ vb, const bf16_t* __restrict__ attn,
            const float* __restrict__ outH, const float* __restrict__ x,
            const float* __restrict__ gamma, float* __restrict__ out) {
  __shared__ __attribute__((aligned(16))) bf16_t Av[128][96];  // [c][v']
  __shared__ __attribute__((aligned(16))) bf16_t Ba[96][96];   // [w][v']
  const int tid = threadIdx.x, lane = tid & 31, wv = tid >> 5;
  const int h = blockIdx.x, cBase = blockIdx.y * 128, b = blockIdx.z;
  const bf16_t* vB = vb + (size_t)b * C_ * N_;

  // V tile: v'-contiguous -> b128 loads + b128 LDS stores (1536 octets, 6/thread)
  #pragma unroll
  for (int it = 0; it < 6; ++it) {
    int od = tid + it * 256;                    // 0..1535
    int ci = od / 12, v8 = (od % 12) * 8;
    *(u32x4*)&Av[ci][v8] =
      *(const u32x4*)(vB + (size_t)(cBase + ci) * N_ + (size_t)h * W_ + v8);
  }
  // attn tile: v'-contiguous -> 8B loads + 8B LDS stores
  #pragma unroll
  for (int it = 0; it < 9; ++it) {
    int qd = tid + it * 256;                    // 0..2303
    int w = qd / 24, v4 = (qd % 24) * 4;
    *(u32x2*)&Ba[w][v4] =
      *(const u32x2*)(attn + ((((size_t)b * H_ + h) * W_ + w) * J_ + H_ + v4));
  }
  __syncthreads();

  const int m0 = wv * 16;
  v8f acc[6];
  #pragma unroll
  for (int t = 0; t < 6; ++t) for (int r = 0; r < 8; ++r) acc[t][r] = 0.f;
  #pragma unroll
  for (int kk = 0; kk < 96; kk += 32) {
    v16bf a = lds_load_a(&Av[0][0], 96, m0, kk, lane);
    #pragma unroll
    for (int t = 0; t < 6; ++t) {
      v16bf bb = lds_load_b(&Ba[0][0], 96, t * 16, kk, lane);
      acc[t] = wmma_bf16(a, bb, acc[t]);
    }
  }
  const float g0 = gamma[0];
  const int half = lane >> 4, n_lo = lane & 15;
  #pragma unroll
  for (int t = 0; t < 6; ++t) {
    int w = t * 16 + n_lo;
    #pragma unroll
    for (int r = 0; r < 8; ++r) {
      int c = cBase + m0 + r + half * 8;
      size_t idx = (((size_t)b * C_ + c) * H_ + h) * W_ + w;
      out[idx] = g0 * (outH[idx] + acc[t][r]) + x[idx];
    }
  }
}

extern "C" void kernel_launch(void* const* d_in, const int* in_sizes, int n_in,
                              void* d_out, int out_size, void* d_ws, size_t ws_size,
                              hipStream_t stream) {
  const float* x     = (const float*)d_in[0];
  const float* Wq    = (const float*)d_in[1];
  const float* bq    = (const float*)d_in[2];
  const float* Wk    = (const float*)d_in[3];
  const float* bk    = (const float*)d_in[4];
  const float* Wv    = (const float*)d_in[5];
  const float* bv    = (const float*)d_in[6];
  const float* gamma = (const float*)d_in[7];
  float* out = (float*)d_out;

  char* ws = (char*)d_ws;
  size_t off = 0;
  auto alloc = [&](size_t bytes) {
    char* p = ws + off;
    off = (off + bytes + 255) & ~(size_t)255;
    return p;
  };
  bf16_t* qb   = (bf16_t*)alloc((size_t)B_ * CQ_ * N_ * sizeof(bf16_t)); //  9.4 MB
  bf16_t* kb   = (bf16_t*)alloc((size_t)B_ * CQ_ * N_ * sizeof(bf16_t)); //  9.4 MB
  bf16_t* vb   = (bf16_t*)alloc((size_t)B_ * C_  * N_ * sizeof(bf16_t)); // 75.5 MB
  bf16_t* Wall = (bf16_t*)alloc((size_t)640 * C_ * sizeof(bf16_t));      //  0.7 MB
  float*  ball = (float*) alloc((size_t)640 * sizeof(float));
  float*  eBuf = (float*) alloc((size_t)B_ * H_ * W_ * J_ * sizeof(float));   // 56.6 MB
  bf16_t* attn = (bf16_t*)alloc((size_t)B_ * H_ * W_ * J_ * sizeof(bf16_t)); // 28.3 MB
  float*  outH = (float*) alloc((size_t)B_ * C_ * N_ * sizeof(float));        // 151 MB

  k_pack<<<dim3((640 * C_ + 255) / 256), 256, 0, stream>>>(Wq, bq, Wk, bk, Wv, bv, Wall, ball);
  k_qkv<<<dim3(N_ / 64, 640 / 64, B_), 128, 0, stream>>>(x, Wall, ball, qb, kb, vb);
  k_energy<0><<<dim3(W_, B_), 192, 0, stream>>>(qb, kb, eBuf);
  k_energy<1><<<dim3(H_, B_), 192, 0, stream>>>(qb, kb, eBuf);
  k_softmax<<<dim3(W_, B_), 192, 0, stream>>>(eBuf, attn);
  k_outH<<<dim3(W_, C_ / 128, B_), 256, 0, stream>>>(vb, attn, outH);
  k_outW<<<dim3(H_, C_ / 128, B_), 256, 0, stream>>>(vb, attn, outH, x, gamma, out);
}
